// AggAttention_80255758893466
// MI455X (gfx1250) — compile-verified, matrix operand order checked by reading, not run on hardware
//
#include <hip/hip_runtime.h>
#include <cmath>

// Problem constants (from reference): K,B,L,D_IN,H = 4,16,4096,512,128
#define KG  4
#define BB  16
#define LL  4096
#define DD  512
#define HH  128
#define TM  8                    // (b,l) rows per workgroup tile
#define RR  (KG * TM)            // 32 flat GEMM rows (row = g*TM + m)
#define NBL (BB * LL)            // 65536 flattened (b,l)
#define DPF (DD + 4)             // padded f32 LDS row (+16B): rotates banks per row
#define DPB (DD + 8)             // padded bf16 LDS row (+16B)

typedef __attribute__((ext_vector_type(16))) __bf16 v16bf;
typedef __attribute__((ext_vector_type(8)))  __bf16 v8bf;
typedef __attribute__((ext_vector_type(4)))  __bf16 v4bf;
typedef __attribute__((ext_vector_type(8)))  float  v8f;

union BF16Frag { v16bf full; v8bf half[2]; };

// ---- CDNA5 feature gates (safe: fall back if toolchain lacks the builtin) ----
#if defined(__has_builtin)
#if __has_builtin(__builtin_amdgcn_global_load_async_to_lds_b128) && \
    __has_builtin(__builtin_amdgcn_s_wait_asynccnt)
#define USE_ASYNC 1
#else
#define USE_ASYNC 0
#endif
#if __has_builtin(__builtin_amdgcn_tanhf)
#define FAST_TANH(x) __builtin_amdgcn_tanhf(x)
#else
#define FAST_TANH(x) tanhf(x)
#endif
#else
#define USE_ASYNC 0
#define FAST_TANH(x) tanhf(x)
#endif

#if USE_ASYNC
// Prototype (from clang diagnostic): params are 'int __vector_size__(16)' ptrs, AS1/AS3
typedef int async_v4i __attribute__((vector_size(4 * sizeof(int))));
typedef __attribute__((address_space(1))) async_v4i ga_v4i;   // global (AS1)
typedef __attribute__((address_space(3))) async_v4i la_v4i;   // LDS (AS3)
#endif

// ---------------------------------------------------------------------------
// Prep: q_bias[h] = W_q[h,:]·vq + ln_b[h]   (128 floats)
//       whb      = bf16(W_h)               (128 x 512)
// ---------------------------------------------------------------------------
__global__ __launch_bounds__(128)
void prep_kernel(const float* __restrict__ ln_w,
                 const float* __restrict__ ln_b,
                 const float* __restrict__ vq,
                 float* __restrict__ qbias,
                 __bf16* __restrict__ whb) {
    const int hrow = threadIdx.x;
    if (hrow >= HH) return;
    const float* wr = ln_w + (size_t)hrow * (DD + HH);
    float acc = ln_b[hrow];
    #pragma unroll 8
    for (int j = 0; j < HH; ++j) acc += wr[DD + j] * vq[j];
    qbias[hrow] = acc;
    #pragma unroll 4
    for (int d = 0; d < DD; ++d) whb[(size_t)hrow * DD + d] = (__bf16)wr[d];
}

// ---------------------------------------------------------------------------
// Main fused kernel: one block handles 8 (b,l) positions x all K=4 groups.
//   Phase 0: async-DMA fp32 hs tile -> LDS (ASYNCcnt), LDS->LDS cvt to bf16
//   Phase 1: bf16 WMMA GEMM 32x512x128 (f32 acc), conversion-free inner loop
//   Phase 2: v_tanh -> v_w dot (shfl+LDS atomics) -> softmax over K
//   Phase 3: x[bl,:] = sum_g a_g * hs_f32[g,bl,:]  (from LDS, single HBM pass)
// ---------------------------------------------------------------------------
__global__ __launch_bounds__(256)
void agg_attn_kernel(const float* __restrict__ hs,
                     const int*   __restrict__ mask,
                     const float* __restrict__ qbias,
                     const __bf16* __restrict__ whb,
                     const float* __restrict__ v_w,
                     float* __restrict__ out) {
    __shared__ float  lds_f[RR][DPF];   // fp32 hs tile, ~64.5 KB, padded rows
    __shared__ __bf16 lds_a[RR][DPB];   // bf16 A tile,  ~32.5 KB, padded rows
    __shared__ float  lds_s[KG][TM];    // per-(k,row) scores -> softmax weights

    const int tid = threadIdx.x;
    const int bl0 = blockIdx.x * TM;

    if (tid < KG * TM) ((float*)lds_s)[tid] = 0.0f;

    // ---- Phase 0a: async-stage hs tile (32 rows x 2 KB) into LDS f32.
    // CDNA5: GLOBAL_LOAD_ASYNC_TO_LDS_B128, tracked by ASYNCcnt (no VGPR roundtrip).
    {
        for (int i = tid; i < RR * (DD / 4); i += 256) {   // 4096 x 16B chunks
            const int row = i >> 7;                        // flat row = g*TM + m
            const int dv  = i & 127;
            const int g   = row >> 3;
            const int m   = row & 7;
            const float* gsrc = hs + ((size_t)g * NBL + (bl0 + m)) * DD + dv * 4;
            float* ldst = &lds_f[row][dv * 4];
#if USE_ASYNC
            __builtin_amdgcn_global_load_async_to_lds_b128(
                (ga_v4i*)gsrc, (la_v4i*)ldst, 0, 0);
#else
            *(float4*)ldst = *(const float4*)gsrc;
#endif
        }
#if USE_ASYNC
        __builtin_amdgcn_s_wait_asynccnt(0);
#endif
    }
    __syncthreads();

    // ---- Phase 0b: one-shot LDS f32 -> LDS bf16 convert (keeps GEMM loop pure)
    {
        for (int i = tid; i < RR * (DD / 4); i += 256) {
            const int row = i >> 7;
            const int dv  = i & 127;
            const float4 v = *(const float4*)(&lds_f[row][dv * 4]);
            *(v4bf*)(&lds_a[row][dv * 4]) =
                (v4bf){(__bf16)v.x, (__bf16)v.y, (__bf16)v.z, (__bf16)v.w};
        }
    }
    __syncthreads();

    // ---- Phase 1: WMMA GEMM. Wave w owns N columns [16w,16w+16);
    // 2 accumulators = M-tiles t (flat rows t*16 .. t*16+15).
    const int wave  = tid >> 5;
    const int lane  = tid & 31;
    const int hhalf = lane >> 4;     // 0/1: 16-lane half (K-split in A/B layout)
    const int nloc  = lane & 15;     // column within wave's N-slice; also A-row id
    const int n     = wave * 16 + nloc;

    v8f acc0 = (v8f){0.f,0.f,0.f,0.f,0.f,0.f,0.f,0.f};
    v8f acc1 = acc0;

    const __bf16* brow = whb + (size_t)n * DD;
    __builtin_prefetch(brow, 0, 3);   // global_prefetch_b8: warm W_h slice

    const __bf16* ar0 = &lds_a[nloc][0];        // M-tile 0: rows 0..15
    const __bf16* ar1 = &lds_a[16 + nloc][0];   // M-tile 1: rows 16..31

    #pragma unroll 4
    for (int kk = 0; kk < DD; kk += 32) {
        // B fragment (32x16 K x N), bf16 from workspace (L2-resident)
        BF16Frag bf;
        bf.half[0] = *(const v8bf*)(brow + kk + 8 * hhalf);
        bf.half[1] = *(const v8bf*)(brow + kk + 16 + 8 * hhalf);

        BF16Frag a0, a1;
        a0.half[0] = *(const v8bf*)(ar0 + kk + 8 * hhalf);
        a0.half[1] = *(const v8bf*)(ar0 + kk + 16 + 8 * hhalf);
        a1.half[0] = *(const v8bf*)(ar1 + kk + 8 * hhalf);
        a1.half[1] = *(const v8bf*)(ar1 + kk + 16 + 8 * hhalf);

        acc0 = __builtin_amdgcn_wmma_f32_16x16x32_bf16(
            false, a0.full, false, bf.full, (short)0, acc0, false, false);
        acc1 = __builtin_amdgcn_wmma_f32_16x16x32_bf16(
            false, a1.full, false, bf.full, (short)0, acc1, false, false);
    }

    // ---- Phase 2a: h = tanh(acc + q_bias[n]); partial s = h * v_w[n]
    // C/D layout: element r of lane -> (flat row = t*16 + r + 8*hhalf, N = nloc)
    {
        const float qb = qbias[n];
        const float vw = v_w[n];
        #pragma unroll
        for (int t = 0; t < 2; ++t) {
            #pragma unroll
            for (int r = 0; r < 8; ++r) {
                const float v = (t == 0) ? acc0[r] : acc1[r];
                float p = FAST_TANH(v + qb) * vw;
                // reduce over the 16 N-columns held by this wave (stays in half)
                p += __shfl_xor(p, 1, 32);
                p += __shfl_xor(p, 2, 32);
                p += __shfl_xor(p, 4, 32);
                p += __shfl_xor(p, 8, 32);
                if (nloc == 0) {
                    const int row = t * 16 + r + 8 * hhalf;
                    atomicAdd(&lds_s[row >> 3][row & 7], p);
                }
            }
        }
    }
    __syncthreads();

    // ---- Phase 2b: mask + softmax over K (threads 0..7, one per row)
    // (mask offset is identical across K so it cancels in softmax; kept for fidelity)
    if (tid < TM) {
        const int m = tid;
        const float off = (mask[bl0 + m] == 0) ? -10000.0f : 0.0f;
        float s0 = lds_s[0][m] + off, s1 = lds_s[1][m] + off;
        float s2 = lds_s[2][m] + off, s3 = lds_s[3][m] + off;
        const float mx = fmaxf(fmaxf(s0, s1), fmaxf(s2, s3));
        const float e0 = __expf(s0 - mx), e1 = __expf(s1 - mx);
        const float e2 = __expf(s2 - mx), e3 = __expf(s3 - mx);
        const float inv = 1.0f / (e0 + e1 + e2 + e3);
        lds_s[0][m] = e0 * inv; lds_s[1][m] = e1 * inv;
        lds_s[2][m] = e2 * inv; lds_s[3][m] = e3 * inv;
    }
    __syncthreads();

    // ---- Phase 3: x[bl,:] = sum_g a[g] * hs[g,bl,:]  (fp32 from LDS)
    {
        const int m = tid >> 5;        // row 0..7
        const int c = tid & 31;        // 32 threads per row
        const float a0 = lds_s[0][m], a1 = lds_s[1][m];
        const float a2 = lds_s[2][m], a3 = lds_s[3][m];
        const float* r0 = &lds_f[0 * TM + m][0];
        const float* r1 = &lds_f[1 * TM + m][0];
        const float* r2 = &lds_f[2 * TM + m][0];
        const float* r3 = &lds_f[3 * TM + m][0];
        const size_t ro = (size_t)(bl0 + m) * DD;
        #pragma unroll
        for (int j = 0; j < 4; ++j) {
            const int dv = (c + j * 32) * 4;
            const float4 x0 = *(const float4*)(r0 + dv);
            const float4 x1 = *(const float4*)(r1 + dv);
            const float4 x2 = *(const float4*)(r2 + dv);
            const float4 x3 = *(const float4*)(r3 + dv);
            float4 o;
            o.x = a0 * x0.x + a1 * x1.x + a2 * x2.x + a3 * x3.x;
            o.y = a0 * x0.y + a1 * x1.y + a2 * x2.y + a3 * x3.y;
            o.z = a0 * x0.z + a1 * x1.z + a2 * x2.z + a3 * x3.z;
            o.w = a0 * x0.w + a1 * x1.w + a2 * x2.w + a3 * x3.w;
            *(float4*)(out + ro + dv) = o;
        }
    }
}

// ---------------------------------------------------------------------------
extern "C" void kernel_launch(void* const* d_in, const int* in_sizes, int n_in,
                              void* d_out, int out_size, void* d_ws, size_t ws_size,
                              hipStream_t stream) {
    const float* hs   = (const float*)d_in[0];
    const int*   mask = (const int*)  d_in[1];
    const float* ln_w = (const float*)d_in[2];
    const float* ln_b = (const float*)d_in[3];
    const float* v_w  = (const float*)d_in[4];
    const float* vq   = (const float*)d_in[5];
    float* out = (float*)d_out;

    // Workspace: [0,512)  q_bias (128 f32); [1024, 1024+128*512*2) bf16 W_h
    float*  qbias = (float*)d_ws;
    __bf16* whb   = (__bf16*)((char*)d_ws + 1024);

    prep_kernel<<<1, 128, 0, stream>>>(ln_w, ln_b, vq, qbias, whb);
    agg_attn_kernel<<<NBL / TM, 256, 0, stream>>>(hs, mask, qbias, whb, v_w, out);
}
